// Attention_7361573946011
// MI455X (gfx1250) — matrix operand order repaired and checked by reading, not compiled, not run on hardware
//
#include <hip/hip_runtime.h>
#include <hip/hip_bf16.h>

typedef __attribute__((ext_vector_type(16))) __bf16 v16bf;
typedef __attribute__((ext_vector_type(8)))  float  v8f;
typedef int v4i_vec __attribute__((vector_size(16)));

constexpr int Bn  = 2;
constexpr int Qn  = 1024;
constexpr int Hn  = 32;
constexpr int KVH = 8;
constexpr int Dh  = 128;
constexpr int KVn = 4096;
constexpr int HID = 4096;
constexpr int Mrows = Bn * Qn;        // 2048

union Frag { v16bf v; uint4 u[2]; };

__device__ __forceinline__ unsigned bfr(float f) {
  union { float f; unsigned u; } x; x.f = f;
  return (x.u + 0x7FFFu + ((x.u >> 16) & 1u)) >> 16;
}
__device__ __forceinline__ __bf16 f2bf(float f) {
  union { unsigned short s; __bf16 b; } y; y.s = (unsigned short)bfr(f);
  return y.b;
}
__device__ __forceinline__ unsigned pack2bf(float lo, float hi) {
  return bfr(lo) | (bfr(hi) << 16);
}

// -------------------- gfx1250 async global->LDS staging ---------------------
#if __has_builtin(__builtin_amdgcn_global_load_async_to_lds_b128) && \
    __has_builtin(__builtin_amdgcn_s_wait_asynccnt)
#define USE_ASYNC_COPY 1
#else
#define USE_ASYNC_COPY 0
#endif

__device__ __forceinline__ void async_copy16(const void* gsrc, void* lds) {
#if USE_ASYNC_COPY
  __builtin_amdgcn_global_load_async_to_lds_b128(
      (__attribute__((address_space(1))) v4i_vec*)gsrc,
      (__attribute__((address_space(3))) v4i_vec*)lds, 0, 0);
#else
  *(uint4*)lds = *(const uint4*)gsrc;
#endif
}
__device__ __forceinline__ void stage_wait() {
#if USE_ASYNC_COPY
  __builtin_amdgcn_s_wait_asynccnt(0);
#endif
  __syncthreads();
}

// ---------------------------------------------------------------------------
// C(M,N) = A(M,K) * W(N,K)^T, all-bf16 operands, fp32 out.
// 256 threads = 8 waves; block tile 128x128; K-step 32; double-buffered LDS
// staged via async global->LDS. Wave w computes rows w*16..w*16+15, cols 0..127.
// All 8 B fragments are preloaded into distinct registers so the 8 WMMAs can
// issue back-to-back instead of being fenced by per-fragment s_wait_dscnt.
// ---------------------------------------------------------------------------
__global__ __launch_bounds__(256) void gemm_bf16_kernel(
    const __bf16* __restrict__ A, const __bf16* __restrict__ W,
    float* __restrict__ C, int M, int N, int K) {
  __shared__ __align__(32) __bf16 As[2][128][32];
  __shared__ __align__(32) __bf16 Ws[2][128][32];
  const int tid  = threadIdx.x;
  const int wave = tid >> 5, lane = tid & 31;
  const int lh = lane >> 4, l16 = lane & 15;
  const int bm = blockIdx.y * 128, bn = blockIdx.x * 128;

  // 512 16-byte chunks per operand tile (128 rows x 4 chunks), 2 per thread
  auto stage = [&](int buf, int k0) {
    #pragma unroll
    for (int t = 0; t < 2; ++t) {
      const int c = tid + t * 256;
      const int r = c >> 2, col = (c & 3) * 8;
      async_copy16(A + (size_t)(bm + r) * K + k0 + col, &As[buf][r][col]);
      async_copy16(W + (size_t)(bn + r) * K + k0 + col, &Ws[buf][r][col]);
    }
  };

  v8f acc[8];
  #pragma unroll
  for (int t = 0; t < 8; ++t) acc[t] = v8f{0.f,0.f,0.f,0.f,0.f,0.f,0.f,0.f};

  stage(0, 0);
  stage_wait();
  int cur = 0;
  for (int k0 = 0; k0 < K; k0 += 32) {
    if (k0 + 32 < K) stage(cur ^ 1, k0 + 32);

    // A fragment: row m = wave*16 + l16; elems 0..7 -> k=lh*8+0..7, 8..15 -> +16
    const int m = wave * 16 + l16;
    Frag fa;
    const __bf16* ap = &As[cur][m][lh * 8];
    fa.u[0] = *reinterpret_cast<const uint4*>(ap);
    fa.u[1] = *reinterpret_cast<const uint4*>(ap + 16);

    // preload all 8 B fragments (distinct registers -> batched DS waits)
    Frag fb[8];
    #pragma unroll
    for (int t = 0; t < 8; ++t) {
      // B fragment: col n = t*16 + l16; elems 0..7 -> k=lh*8+0..7, 8..15 -> +16
      const __bf16* bp = &Ws[cur][t * 16 + l16][lh * 8];
      fb[t].u[0] = *reinterpret_cast<const uint4*>(bp);
      fb[t].u[1] = *reinterpret_cast<const uint4*>(bp + 16);
    }
    #pragma unroll
    for (int t = 0; t < 8; ++t) {
      acc[t] = __builtin_amdgcn_wmma_f32_16x16x32_bf16(
          false, fa.v, false, fb[t].v, (short)0, acc[t], false, false);
    }
    stage_wait();   // staged tile landed; all waves done reading As[cur]
    cur ^= 1;
  }

  #pragma unroll
  for (int t = 0; t < 8; ++t) {
    const int n = bn + t * 16 + l16;
    #pragma unroll
    for (int j = 0; j < 8; ++j) {
      const int mr = bm + wave * 16 + lh * 8 + j;
      C[(size_t)mr * N + n] = acc[t][j];
    }
  }
}

// ---------------------------------------------------------------------------
// fp32 -> bf16 bulk convert, 8 elements/thread, packed dword stores
// ---------------------------------------------------------------------------
__global__ void f32_to_bf16_kernel(const float* __restrict__ src,
                                   __bf16* __restrict__ dst) {
  const size_t i = ((size_t)blockIdx.x * blockDim.x + threadIdx.x) * 8;
  const float4 a = *reinterpret_cast<const float4*>(src + i);
  const float4 b = *reinterpret_cast<const float4*>(src + i + 4);
  uint4 o;
  o.x = pack2bf(a.x, a.y);
  o.y = pack2bf(a.z, a.w);
  o.z = pack2bf(b.x, b.y);
  o.w = pack2bf(b.z, b.w);
  *reinterpret_cast<uint4*>(dst + i) = o;
}

// cache_v (b,kvh,kv,d) fp32 -> Vt bf16 transposed (b,kvh,d,kv)
__global__ void conv_v_kernel(const float* __restrict__ cv, __bf16* __restrict__ Vt) {
  size_t i = (size_t)blockIdx.x * blockDim.x + threadIdx.x;
  const int d  = (int)(i % Dh);  size_t r = i / Dh;
  const int kv = (int)(r % KVn); const size_t bh = r / KVn;  // b*KVH + kvh
  Vt[(bh * Dh + d) * (size_t)KVn + kv] = f2bf(cv[i]);
}

// ---------------------------------------------------------------------------
// RoPE on Q: Qraw (b*q, H*Dh) fp32 -> Qb bf16 (b,h,q,d)
// ---------------------------------------------------------------------------
__global__ void rope_q_kernel(const float* __restrict__ Qraw,
                              const float* __restrict__ cosp,
                              const float* __restrict__ sinp,
                              __bf16* __restrict__ Qb) {
  size_t i = (size_t)blockIdx.x * blockDim.x + threadIdx.x;  // over Bn*Hn*Qn*Dh
  const int d = (int)(i % Dh); size_t r = i / Dh;
  const int q = (int)(r % Qn); r /= Qn;
  const int h = (int)(r % Hn); const int b = (int)(r / Hn);
  const size_t row = (size_t)b * Qn + q;
  const size_t base = row * (Hn * Dh) + h * Dh;
  const float v = Qraw[base + d];
  const float p = (d < Dh / 2) ? -Qraw[base + d + Dh / 2] : Qraw[base + d - Dh / 2];
  const float c = cosp[row * Dh + d], s = sinp[row * Dh + d];
  Qb[i] = f2bf(v * c + p * s);
}

// RoPE on K + scatter K into Kc (row sink_ids[q]) and V into Vt (column)
__global__ void rope_kv_scatter_kernel(const float* __restrict__ Kraw,
                                       const float* __restrict__ Vraw,
                                       const float* __restrict__ cosp,
                                       const float* __restrict__ sinp,
                                       const int* __restrict__ sink,
                                       __bf16* __restrict__ Kc,
                                       __bf16* __restrict__ Vt) {
  size_t i = (size_t)blockIdx.x * blockDim.x + threadIdx.x;  // over Bn*KVH*Qn*Dh
  const int d  = (int)(i % Dh); size_t r = i / Dh;
  const int q  = (int)(r % Qn); r /= Qn;
  const int kh = (int)(r % KVH); const int b = (int)(r / KVH);
  const size_t row = (size_t)b * Qn + q;
  const size_t base = row * (KVH * Dh) + kh * Dh;
  const float kv = Kraw[base + d];
  const float kp = (d < Dh / 2) ? -Kraw[base + d + Dh / 2] : Kraw[base + d - Dh / 2];
  const float c = cosp[row * Dh + d], s = sinp[row * Dh + d];
  const int kvrow = sink[q];
  const size_t bh = (size_t)b * KVH + kh;
  Kc[(bh * KVn + kvrow) * Dh + d] = f2bf(kv * c + kp * s);
  Vt[(bh * Dh + d) * (size_t)KVn + kvrow] = f2bf(Vraw[base + d]);
}

// ---------------------------------------------------------------------------
// Flash attention: grid (Qn/64, Hn, Bn), 128 threads = 4 waves x 16 query rows.
// KV consumed in steps of 32; online softmax; bf16 WMMA for QK^T and PV.
// Output written directly in bf16 for the output-projection GEMM.
// ---------------------------------------------------------------------------
__global__ __launch_bounds__(128) void flash_attn_kernel(
    const __bf16* __restrict__ Qb, const __bf16* __restrict__ Kc,
    const __bf16* __restrict__ Vt, const float* __restrict__ mask,
    __bf16* __restrict__ Out) {
  const int qblk = blockIdx.x, h = blockIdx.y, b = blockIdx.z;
  const int wave = threadIdx.x >> 5, lane = threadIdx.x & 31;
  const int lh = lane >> 4, l16 = lane & 15;
  const int kh = h / (Hn / KVH);
  const int q0 = qblk * 64 + wave * 16;
  const __bf16* Qp = Qb + ((size_t)(b * Hn + h) * Qn + q0) * Dh;
  const __bf16* Kp = Kc + (size_t)(b * KVH + kh) * KVn * Dh;
  const __bf16* Vp = Vt + (size_t)(b * KVH + kh) * Dh * (size_t)KVn;
  const float*  Mp = mask + ((size_t)b * Qn + q0) * KVn;
  __shared__ __align__(32) __bf16 Plds[4][16][32];

  // Q fragments (16 rows x 128 d) held for the whole KV loop
  v16bf qf[4];
  #pragma unroll
  for (int kb = 0; kb < 4; ++kb) {
    const __bf16* src = Qp + (size_t)l16 * Dh + kb * 32 + lh * 8;
    Frag f;
    f.u[0] = *reinterpret_cast<const uint4*>(src);
    f.u[1] = *reinterpret_cast<const uint4*>(src + 16);
    qf[kb] = f.v;
  }

  v8f acc[8];
  #pragma unroll
  for (int t = 0; t < 8; ++t) acc[t] = v8f{0.f,0.f,0.f,0.f,0.f,0.f,0.f,0.f};
  float rm[8], lsum[8];
  #pragma unroll
  for (int j = 0; j < 8; ++j) { rm[j] = -3.0e38f; lsum[j] = 0.f; }
  const float sc = 0.08838834764831845f;  // 1/sqrt(128)

  for (int kv0 = 0; kv0 < KVn; kv0 += 32) {
    v8f s0 = v8f{0.f,0.f,0.f,0.f,0.f,0.f,0.f,0.f};
    v8f s1 = v8f{0.f,0.f,0.f,0.f,0.f,0.f,0.f,0.f};
    #pragma unroll
    for (int kb = 0; kb < 4; ++kb) {
      const __bf16* k0p = Kp + (size_t)(kv0 + l16) * Dh + kb * 32 + lh * 8;
      Frag kf0; kf0.u[0] = *reinterpret_cast<const uint4*>(k0p);
      kf0.u[1] = *reinterpret_cast<const uint4*>(k0p + 16);
      const __bf16* k1p = Kp + (size_t)(kv0 + 16 + l16) * Dh + kb * 32 + lh * 8;
      Frag kf1; kf1.u[0] = *reinterpret_cast<const uint4*>(k1p);
      kf1.u[1] = *reinterpret_cast<const uint4*>(k1p + 16);
      s0 = __builtin_amdgcn_wmma_f32_16x16x32_bf16(false, qf[kb], false, kf0.v,
                                                   (short)0, s0, false, false);
      s1 = __builtin_amdgcn_wmma_f32_16x16x32_bf16(false, qf[kb], false, kf1.v,
                                                   (short)0, s1, false, false);
    }

    // online softmax over the 32 new columns (row = lh*8 + j, cols across lanes)
    #pragma unroll
    for (int j = 0; j < 8; ++j) {
      const int m = lh * 8 + j;
      float sm = sc * s0[j] + Mp[(size_t)m * KVn + kv0 + l16];
      float sn = sc * s1[j] + Mp[(size_t)m * KVn + kv0 + 16 + l16];
      float tmax = fmaxf(sm, sn);
      #pragma unroll
      for (int off = 8; off > 0; off >>= 1)
        tmax = fmaxf(tmax, __shfl_xor(tmax, off, 32));
      const float mnew  = fmaxf(rm[j], tmax);
      const float alpha = __expf(rm[j] - mnew);
      rm[j] = mnew;
      const float p0 = __expf(sm - mnew);
      const float p1 = __expf(sn - mnew);
      float rs = p0 + p1;
      #pragma unroll
      for (int off = 8; off > 0; off >>= 1)
        rs += __shfl_xor(rs, off, 32);
      lsum[j] = lsum[j] * alpha + rs;
      #pragma unroll
      for (int t = 0; t < 8; ++t) acc[t][j] *= alpha;
      Plds[wave][m][l16]      = f2bf(p0);
      Plds[wave][m][16 + l16] = f2bf(p1);
    }

    // read P back in A-matrix layout (per-wave private LDS region; DS in-order)
    Frag pf;
    const __bf16* pr = &Plds[wave][l16][lh * 8];
    pf.u[0] = *reinterpret_cast<const uint4*>(pr);
    pf.u[1] = *reinterpret_cast<const uint4*>(pr + 16);

    // acc += P(16x32) * V(32x16 per d-tile); Vt is kv-contiguous -> 32B loads
    #pragma unroll
    for (int t = 0; t < 8; ++t) {
      const __bf16* vp = Vp + (size_t)(t * 16 + l16) * KVn + kv0 + lh * 8;
      Frag vf; vf.u[0] = *reinterpret_cast<const uint4*>(vp);
      vf.u[1] = *reinterpret_cast<const uint4*>(vp + 16);
      acc[t] = __builtin_amdgcn_wmma_f32_16x16x32_bf16(false, pf.v, false, vf.v,
                                                       (short)0, acc[t], false, false);
    }
  }

  // normalize and write bf16 (B, Q, H*D)
  #pragma unroll
  for (int j = 0; j < 8; ++j) {
    const int m = lh * 8 + j;
    const float inv = 1.0f / lsum[j];
    #pragma unroll
    for (int t = 0; t < 8; ++t) {
      Out[((size_t)b * Qn + q0 + m) * (Hn * Dh) + h * Dh + t * 16 + l16] =
          f2bf(acc[t][j] * inv);
    }
  }
}

// ---------------------------------------------------------------------------
extern "C" void kernel_launch(void* const* d_in, const int* in_sizes, int n_in,
                              void* d_out, int out_size, void* d_ws, size_t ws_size,
                              hipStream_t stream) {
  (void)in_sizes; (void)n_in; (void)out_size; (void)ws_size;
  const float* hidden  = (const float*)d_in[0];
  const float* cosp    = (const float*)d_in[1];
  const float* sinp    = (const float*)d_in[2];
  const float* mask    = (const float*)d_in[3];
  const float* cache_k = (const float*)d_in[4];
  const float* cache_v = (const float*)d_in[5];
  const int*   sink    = (const int*)d_in[6];
  const float* Wq      = (const float*)d_in[7];
  const float* Wk      = (const float*)d_in[8];
  const float* Wv      = (const float*)d_in[9];
  const float* Wo      = (const float*)d_in[10];
  float* out = (float*)d_out;

  char* p = (char*)d_ws;
  auto alloc = [&](size_t bytes) {
    char* r = p; p += (bytes + 255) & ~(size_t)255; return r;
  };
  float*  Qraw  = (float*)alloc((size_t)Mrows * (Hn * Dh) * sizeof(float));
  float*  Kraw  = (float*)alloc((size_t)Mrows * (KVH * Dh) * sizeof(float));
  float*  Vraw  = (float*)alloc((size_t)Mrows * (KVH * Dh) * sizeof(float));
  __bf16* Ah    = (__bf16*)alloc((size_t)Mrows * HID * sizeof(__bf16));
  __bf16* Wqb   = (__bf16*)alloc((size_t)(Hn * Dh) * HID * sizeof(__bf16));
  __bf16* Wkb   = (__bf16*)alloc((size_t)(KVH * Dh) * HID * sizeof(__bf16));
  __bf16* Wvb   = (__bf16*)alloc((size_t)(KVH * Dh) * HID * sizeof(__bf16));
  __bf16* Wob   = (__bf16*)alloc((size_t)HID * (Hn * Dh) * sizeof(__bf16));
  __bf16* attnb = (__bf16*)alloc((size_t)Mrows * (Hn * Dh) * sizeof(__bf16));
  __bf16* Qb    = (__bf16*)alloc((size_t)Bn * Hn * Qn * Dh * sizeof(__bf16));
  __bf16* Kc    = (__bf16*)alloc((size_t)Bn * KVH * KVn * Dh * sizeof(__bf16));
  __bf16* Vt    = (__bf16*)alloc((size_t)Bn * KVH * KVn * Dh * sizeof(__bf16));

  auto cvt = [&](const float* s, __bf16* d, size_t n) {
    f32_to_bf16_kernel<<<dim3((unsigned)(n / 8 / 256)), 256, 0, stream>>>(s, d);
  };

  // 1) bulk fp32 -> bf16 conversions (one pass each; all fit in 192MB L2)
  cvt(hidden,  Ah,  (size_t)Mrows * HID);
  cvt(Wq,      Wqb, (size_t)(Hn * Dh) * HID);
  cvt(Wk,      Wkb, (size_t)(KVH * Dh) * HID);
  cvt(Wv,      Wvb, (size_t)(KVH * Dh) * HID);
  cvt(Wo,      Wob, (size_t)HID * (Hn * Dh));
  cvt(cache_k, Kc,  (size_t)Bn * KVH * KVn * Dh);
  conv_v_kernel<<<dim3((unsigned)((size_t)Bn * KVH * KVn * Dh / 256)), 256, 0, stream>>>(
      cache_v, Vt);

  // 2) QKV projections (bf16 WMMA GEMMs, async-LDS staged)
  gemm_bf16_kernel<<<dim3(HID / 128, Mrows / 128), 256, 0, stream>>>(
      Ah, Wqb, Qraw, Mrows, Hn * Dh, HID);
  gemm_bf16_kernel<<<dim3((KVH * Dh) / 128, Mrows / 128), 256, 0, stream>>>(
      Ah, Wkb, Kraw, Mrows, KVH * Dh, HID);
  gemm_bf16_kernel<<<dim3((KVH * Dh) / 128, Mrows / 128), 256, 0, stream>>>(
      Ah, Wvb, Vraw, Mrows, KVH * Dh, HID);

  // 3) RoPE + cache scatter at sink_ids
  rope_q_kernel<<<dim3((unsigned)((size_t)Bn * Hn * Qn * Dh / 256)), 256, 0, stream>>>(
      Qraw, cosp, sinp, Qb);
  rope_kv_scatter_kernel<<<dim3((unsigned)((size_t)Bn * KVH * Qn * Dh / 256)), 256, 0, stream>>>(
      Kraw, Vraw, cosp, sinp, sink, Kc, Vt);

  // 4) flash attention over the full cache
  flash_attn_kernel<<<dim3(Qn / 64, Hn, Bn), 128, 0, stream>>>(Qb, Kc, Vt, mask, attnb);

  // 5) output projection (bf16 WMMA GEMM) -> fp32 final output
  gemm_bf16_kernel<<<dim3(HID / 128, Mrows / 128), 256, 0, stream>>>(
      attnb, Wob, out, Mrows, HID, Hn * Dh);
}
